// CRF_45664092291089
// MI455X (gfx1250) — compile-verified, matrix operand order checked by reading, not run on hardware
//
#include <hip/hip_runtime.h>
#include <stdint.h>

#define L_ 512
#define B_ 64
#define C_ 512

typedef __attribute__((ext_vector_type(16))) __bf16 v16bf;
typedef __attribute__((ext_vector_type(8)))  float  v8f;

union Frag { v16bf v; uint4 q[2]; };

__device__ __forceinline__ unsigned short f2bf(float f) {
  unsigned u = __float_as_uint(f);
  unsigned r = (u + 0x7FFFu + ((u >> 16) & 1u)) >> 16;   // round-to-nearest-even
  return (unsigned short)r;
}

// ---------------------------------------------------------------------------
// Preprocess 1: column max of trans
// ---------------------------------------------------------------------------
__global__ void crf_colmax(const float* __restrict__ trans, float* __restrict__ tcol) {
  int j = blockIdx.x * blockDim.x + threadIdx.x;
  if (j >= C_) return;
  float m = -INFINITY;
  for (int i = 0; i < C_; ++i) m = fmaxf(m, trans[(size_t)i * C_ + j]);
  tcol[j] = m;
}

// ---------------------------------------------------------------------------
// Preprocess 2: pack E = bf16(exp(trans - tcolmax)) in WMMA-B fragment order.
// 16-bit B fragment (32x16, wave32): lane = 16*half + n holds column n,
// element e: K = (e<8) ? 8*half+e : 16+8*half+(e-8).
// Packed address: ((nb*16 + kb)*32 + lane)*16 + e   (ushort units)
// ---------------------------------------------------------------------------
__global__ void crf_pack(const float* __restrict__ trans, const float* __restrict__ tcol,
                         unsigned short* __restrict__ Ep) {
  int nb = blockIdx.x;            // 0..31  (N blocks of 16)
  int kb = blockIdx.y;            // 0..15  (K blocks of 32)
  int lane = threadIdx.x;         // 0..31
  int n = lane & 15, half = lane >> 4;
  int gj = nb * 16 + n;
  float tc = tcol[gj];
  size_t base = ((size_t)(nb * 16 + kb) * 32 + lane) * 16;
#pragma unroll
  for (int e = 0; e < 16; ++e) {
    int K  = (e < 8) ? (8 * half + e) : (16 + 8 * half + (e - 8));
    int gi = kb * 32 + K;
    Ep[base + e] = f2bf(__expf(trans[(size_t)gi * C_ + gj] - tc));
  }
}

// ---------------------------------------------------------------------------
// Persistent single-workgroup scan kernel: 1024 threads = 32 waves on one WGP.
// LDS (dynamic, 205056 B)
// ---------------------------------------------------------------------------
#define PA 520
#define PF 516
#define OFF_ABF     0
#define OFF_ALPHA   66560
#define OFF_TCOL    (66560 + 132096)
#define OFF_ROWMAX  (66560 + 132096 + 2048)
#define OFF_SCRATCH (66560 + 132096 + 2048 + 256)
#define SMEM_BYTES  (66560 + 132096 + 2048 + 256 + 4096)

__device__ __forceinline__ void reduce_rowmax(const float* __restrict__ alpha,
                                              float* __restrict__ rowmax,
                                              float* __restrict__ scratch, int tid) {
  int r = tid >> 4, cc = tid & 15;
  const float* row = alpha + r * PF + cc * 32;
  float m = -INFINITY;
#pragma unroll 8
  for (int k = 0; k < 32; ++k) m = fmaxf(m, row[k]);
  scratch[tid] = m;                       // scratch[r*16+cc]
  __syncthreads();
  if (tid < 64) {
    float mm = -INFINITY;
#pragma unroll
    for (int k = 0; k < 16; ++k) mm = fmaxf(mm, scratch[tid * 16 + k]);
    rowmax[tid] = mm;
  }
  __syncthreads();
}

__device__ __forceinline__ void build_A(const float* __restrict__ alpha,
                                        const float* __restrict__ rowmax,
                                        unsigned short* __restrict__ Abf, int tid) {
  int r = tid >> 4, c0 = (tid & 15) * 32;
  float mb = rowmax[r];
  const float* row = alpha + r * PF + c0;
  unsigned short* arow = Abf + r * PA + c0;
#pragma unroll 8
  for (int k = 0; k < 32; ++k) arow[k] = f2bf(__expf(row[k] - mb));
  __syncthreads();
}

__global__ __launch_bounds__(1024) void crf_scan(
    const float* __restrict__ emit, const unsigned char* __restrict__ mask,
    const float* __restrict__ tfs, const float* __restrict__ tte,
    const float* __restrict__ tcol_g, const unsigned short* __restrict__ Ep,
    float* __restrict__ logz_out) {
  extern __shared__ char smem[];
  unsigned short* Abf   = (unsigned short*)(smem + OFF_ABF);
  float* alpha   = (float*)(smem + OFF_ALPHA);
  float* tcol    = (float*)(smem + OFF_TCOL);
  float* rowmax  = (float*)(smem + OFF_ROWMAX);
  float* scratch = (float*)(smem + OFF_SCRATCH);

  const int tid  = threadIdx.x;
  const int lane = tid & 31;
  const int w    = tid >> 5;     // wave id == N block (16 output columns)
  const int n    = lane & 15;
  const int half = lane >> 4;
  const int j    = w * 16 + n;

  // --- init: tcol cache + alpha0 = emit[0] + trans_from_start ---
  for (int i = tid; i < C_; i += 1024) tcol[i] = tcol_g[i];
  for (int i = tid; i < B_ * C_; i += 1024) {
    int b = i >> 9, c = i & (C_ - 1);
    alpha[b * PF + c] = emit[(size_t)b * C_ + c] + tfs[c];
  }
  __syncthreads();
  reduce_rowmax(alpha, rowmax, scratch, tid);
  build_A(alpha, rowmax, Abf, tid);

  // --- scan over time ---
  for (int step = 1; step < L_; ++step) {
    // Opaque zero offset: keeps the E loads inside the step loop (no hoist /
    // scratch spill) while PRESERVING the global-pointer provenance of Ep so
    // the backend emits global_load_b128 (LOADcnt only), not flat_load_b128
    // (which would also bump DScnt and couple against the A-tile ds_loads).
    unsigned int zoff = 0;
    asm volatile("" : "+v"(zoff));
    const unsigned short* epw = Ep + (size_t)w * 16 * 32 * 16 + zoff;
    const uint4* bp = (const uint4*)epw;

    // Phase 1: GEMM  S(64x512) = A(64x512,bf16) x E(512x512,bf16), per-wave
    // N block of 16 columns, 4 M tiles of 16 rows each.
    v8f acc[4] = {};
#pragma unroll
    for (int kb = 0; kb < 16; ++kb) {
      Frag bf;
      bf.q[0] = bp[kb * 64 + lane * 2];
      bf.q[1] = bp[kb * 64 + lane * 2 + 1];
      if (kb < 15) {  // pull next K block toward the near caches
        __builtin_prefetch((const char*)epw + (((kb + 1) * 32 + lane) * 32), 0, 3);
      }
#pragma unroll
      for (int t = 0; t < 4; ++t) {
        Frag af;
        const unsigned short* abase = Abf + (t * 16 + n) * PA + kb * 32 + 8 * half;
        af.q[0] = *(const uint4*)(abase);
        af.q[1] = *(const uint4*)(abase + 16);
        acc[t] = __builtin_amdgcn_wmma_f32_16x16x32_bf16(
            false, af.v, false, bf.v, (short)0, acc[t], false, false);
      }
    }

    // Phase 2: update alpha where mask true:
    //   alpha[b,j] = rowmax_old[b] + tcol[j] + log(S[b,j]) + emit[step,b,j]
    const unsigned char* mrow = mask + (size_t)step * B_;
    const float* erow = emit + (size_t)step * B_ * C_;
    float tc = tcol[j];
#pragma unroll
    for (int t = 0; t < 4; ++t) {
      unsigned long long mbits =
          *(const unsigned long long*)(mrow + t * 16 + half * 8);
#pragma unroll
      for (int v = 0; v < 8; ++v) {
        int b = t * 16 + half * 8 + v;
        if ((mbits >> (8 * v)) & 0xFF) {
          float na = rowmax[b] + tc + __logf(acc[t][v]) + erow[(size_t)b * C_ + j];
          alpha[b * PF + j] = na;
        }
      }
    }
    __syncthreads();

    // Phase 3/4: new row max, rebuild bf16 A = exp(alpha - rowmax)
    reduce_rowmax(alpha, rowmax, scratch, tid);
    build_A(alpha, rowmax, Abf, tid);
  }

  // --- finalize: logZ = sum_b logsumexp_j(alpha[b,:] + tte[:]) ---
  {
    int r = tid >> 4, c0 = (tid & 15) * 32;
    const float* row = alpha + r * PF + c0;
    float m = -INFINITY;
    for (int k = 0; k < 32; ++k) m = fmaxf(m, row[k] + tte[c0 + k]);
    scratch[tid] = m;
    __syncthreads();
    if (tid < 64) {
      float mm = -INFINITY;
      for (int k = 0; k < 16; ++k) mm = fmaxf(mm, scratch[tid * 16 + k]);
      rowmax[tid] = mm;
    }
    __syncthreads();
    float mb = rowmax[r];
    float s = 0.f;
    for (int k = 0; k < 32; ++k) s += __expf(row[k] + tte[c0 + k] - mb);
    __syncthreads();   // everyone done reading rowmax/scratch before rewrite
    scratch[tid] = s;
    __syncthreads();
    if (tid < 64) {
      float ss = 0.f;
      for (int k = 0; k < 16; ++k) ss += scratch[tid * 16 + k];
      rowmax[tid] = rowmax[tid] + __logf(ss);
    }
    __syncthreads();
    if (tid == 0) {
      float tot = 0.f;
      for (int b = 0; b < B_; ++b) tot += rowmax[b];
      *logz_out = tot;
    }
  }
}

// ---------------------------------------------------------------------------
// Score + final output kernel (small): out = (logZ - score)/B
// ---------------------------------------------------------------------------
__global__ __launch_bounds__(1024) void crf_score(
    const float* __restrict__ emit, const int* __restrict__ target,
    const unsigned char* __restrict__ mask, const float* __restrict__ trans,
    const float* __restrict__ tfs, const float* __restrict__ tte,
    const float* __restrict__ logz, float* __restrict__ out) {
  __shared__ float red[1024];
  __shared__ float red2[B_];
  int tid = threadIdx.x;
  float p = 0.f;
  for (int idx = tid; idx < L_ * B_; idx += 1024) {
    int l = idx >> 6;                    // B_ == 64
    if (mask[idx]) {
      int tg = target[idx];
      float s = emit[(size_t)idx * C_ + tg];
      if (l > 0) s += trans[(size_t)target[idx - B_] * C_ + tg];
      p += s;
    }
  }
  red[tid] = p;
  __syncthreads();
  for (int off = 512; off > 0; off >>= 1) {
    if (tid < off) red[tid] += red[tid + off];
    __syncthreads();
  }
  if (tid < B_) {
    int b = tid, cnt = 0;
    for (int l = 0; l < L_; ++l) cnt += mask[l * B_ + b] ? 1 : 0;
    int last = target[(cnt - 1) * B_ + b];
    red2[b] = tfs[target[b]] + tte[last];
  }
  __syncthreads();
  if (tid == 0) {
    float extra = 0.f;
    for (int b = 0; b < B_; ++b) extra += red2[b];
    out[0] = (logz[0] - (red[0] + extra)) / (float)B_;
  }
}

// ---------------------------------------------------------------------------
// Launch
// ---------------------------------------------------------------------------
extern "C" void kernel_launch(void* const* d_in, const int* in_sizes, int n_in,
                              void* d_out, int out_size, void* d_ws, size_t ws_size,
                              hipStream_t stream) {
  const float*         emit   = (const float*)d_in[0];
  const int*           target = (const int*)d_in[1];
  const unsigned char* mask   = (const unsigned char*)d_in[2];   // jnp bool_
  const float*         trans  = (const float*)d_in[3];
  const float*         tfs    = (const float*)d_in[4];
  const float*         tte    = (const float*)d_in[5];

  char* ws = (char*)d_ws;
  float*          tcol = (float*)ws;                         // 2048 B
  unsigned short* Ep   = (unsigned short*)(ws + 2048);       // 512 KB packed E
  float*          logz = (float*)(ws + 2048 + (size_t)C_ * C_ * 2);

  crf_colmax<<<dim3((C_ + 255) / 256), dim3(256), 0, stream>>>(trans, tcol);
  crf_pack<<<dim3(C_ / 16, C_ / 32), dim3(32), 0, stream>>>(trans, tcol, Ep);
  crf_scan<<<dim3(1), dim3(1024), SMEM_BYTES, stream>>>(emit, mask, tfs, tte,
                                                        tcol, Ep, logz);
  crf_score<<<dim3(1), dim3(1024), 0, stream>>>(emit, target, mask, trans,
                                                tfs, tte, logz, (float*)d_out);
}
